// MOEPointWiseFeedForward_27642409517785
// MI455X (gfx1250) — matrix operand order repaired and checked by reading, not compiled, not run on hardware
//
#include <hip/hip_runtime.h>

#define Hc 1024
#define Sc 256
#define Ec 8
#define HS (Hc + 8)                 // padded Htile row stride (elements)
#define MT 32                       // tokens per block tile
#define HT_BYTES ((size_t)MT * HS * 2)
#define SMEM_BYTES (HT_BYTES + MT * sizeof(int))

typedef __attribute__((ext_vector_type(16))) __bf16 v16bf;
typedef __attribute__((ext_vector_type(8)))  __bf16 v8bf;
typedef __attribute__((ext_vector_type(4)))  __bf16 v4bf;
typedef __attribute__((ext_vector_type(8)))  float  v8f;

// ---------------- fp32 -> bf16 elementwise convert (x) ----------------
__global__ __launch_bounds__(256) void moe_cvt_bf16_kernel(
    const float* __restrict__ in, __bf16* __restrict__ out, int n)
{
    const int i = ((int)blockIdx.x * 256 + (int)threadIdx.x) * 4;
    if (i + 3 < n) {
        const float4 v = *(const float4*)(in + i);
        v4bf o;
        o[0] = (__bf16)v.x; o[1] = (__bf16)v.y;
        o[2] = (__bf16)v.z; o[3] = (__bf16)v.w;
        *(v4bf*)(out + i) = o;
    }
}

// ------------- fp32 [K][N] -> bf16 [N][K] tiled transpose-convert -------------
__global__ __launch_bounds__(256) void moe_transpose_cvt_kernel(
    const float* __restrict__ W, __bf16* __restrict__ Wt)
{
    __shared__ float tile[32][33];
    const int mat = blockIdx.z;
    const int k0 = (int)blockIdx.y * 32, n0 = (int)blockIdx.x * 32;
    const float* Wm = W + (size_t)mat * Hc * Hc;
    __bf16* Wtm = Wt + (size_t)mat * Hc * Hc;
    const int tx = threadIdx.x & 31, ty = threadIdx.x >> 5;   // 32 x 8
    #pragma unroll
    for (int j = 0; j < 32; j += 8)
        tile[ty + j][tx] = Wm[(size_t)(k0 + ty + j) * Hc + n0 + tx];
    __syncthreads();
    #pragma unroll
    for (int j = 0; j < 32; j += 8)
        Wtm[(size_t)(n0 + ty + j) * Hc + k0 + tx] = (__bf16)tile[tx][ty + j];
}

// ---------------- Router: routes[b] = argmax(relu(ue@sw1+sb1)@sw2+sb2) ----------------
__global__ __launch_bounds__(256) void moe_router_kernel(
    const float* __restrict__ ue, const float* __restrict__ sw1,
    const float* __restrict__ sb1, const float* __restrict__ sw2,
    const float* __restrict__ sb2, int* __restrict__ routes)
{
    __shared__ float ubuf[8][Hc];
    const int tid  = threadIdx.x;
    const int wave = tid >> 5;
    const int lane = tid & 31;
    const int token = blockIdx.x * 8 + wave;

    const float* urow = ue + (size_t)token * Hc;
    for (int k = lane; k < Hc; k += 32) ubuf[wave][k] = urow[k];
    __syncthreads();

    float h[8];
    #pragma unroll
    for (int j = 0; j < 8; ++j) {
        const int s = j * 32 + lane;
        float acc = sb1[s];
        for (int k = 0; k < Hc; ++k)
            acc = fmaf(ubuf[wave][k], sw1[(size_t)k * Sc + s], acc);
        h[j] = fmaxf(acc, 0.0f);
    }

    float lg[Ec];
    #pragma unroll
    for (int e = 0; e < Ec; ++e) lg[e] = 0.0f;
    #pragma unroll
    for (int j = 0; j < 8; ++j) {
        const int s = j * 32 + lane;
        #pragma unroll
        for (int e = 0; e < Ec; ++e)
            lg[e] = fmaf(h[j], sw2[(size_t)s * Ec + e], lg[e]);
    }
    #pragma unroll
    for (int off = 16; off > 0; off >>= 1)
        #pragma unroll
        for (int e = 0; e < Ec; ++e)
            lg[e] += __shfl_xor(lg[e], off, 32);

    if (lane == 0) {
        int best = 0;
        float bv = lg[0] + sb2[0];
        #pragma unroll
        for (int e = 1; e < Ec; ++e) {
            const float v = lg[e] + sb2[e];
            if (v > bv) { bv = v; best = e; }
        }
        routes[token] = best;
    }
}

// ---------------- Bucketize: offsets[E+1], perm[B] ----------------
__global__ __launch_bounds__(256) void moe_bucketize_kernel(
    const int* __restrict__ routes, int* __restrict__ offsets,
    int* __restrict__ perm, int B)
{
    __shared__ int cnt[Ec];
    __shared__ int cur[Ec];
    const int tid = threadIdx.x;
    if (tid < Ec) cnt[tid] = 0;
    __syncthreads();
    for (int b = tid; b < B; b += 256) atomicAdd(&cnt[routes[b]], 1);
    __syncthreads();
    if (tid == 0) {
        int acc = 0;
        #pragma unroll
        for (int e = 0; e < Ec; ++e) { offsets[e] = acc; cur[e] = acc; acc += cnt[e]; }
        offsets[Ec] = acc;
    }
    __syncthreads();
    for (int b = tid; b < B; b += 256) {
        const int pos = atomicAdd(&cur[routes[b]], 1);
        perm[pos] = b;
    }
}

static __device__ inline v16bf make_frag(v8bf lo, v8bf hi) {
    return __builtin_shufflevector(lo, hi, 0, 1, 2, 3, 4, 5, 6, 7,
                                           8, 9, 10, 11, 12, 13, 14, 15);
}

// ---------------- Fused two-layer FFN: 32-token tile, 16 waves (2M x 8N) ----------------
// Expert mode (perm != nullptr): blockIdx.y = expert, gathered tokens, '=' store.
// User  mode (perm == nullptr): identity tokens, '+=' store.
// Dynamic LDS: Htile[MT][HS] bf16 (66 KB) + tok[MT] int.
__global__ __launch_bounds__(512) void moe_ffn_wmma_kernel(
    const __bf16* __restrict__ Xb,       // [B,Hc] bf16
    const __bf16* __restrict__ W1t,      // [(E),Hc,Hc] bf16, (n,k) layout
    const float*  __restrict__ B1base,
    const __bf16* __restrict__ W2t,      // [(E),Hc,Hc] bf16, (n,k) layout
    const float*  __restrict__ B2base,
    float* __restrict__ Out,
    const int* __restrict__ perm, const int* __restrict__ offsets,
    int accumulate)
{
    extern __shared__ __align__(16) char smem[];
    __bf16* Htile = (__bf16*)smem;                 // [MT][HS]
    int*    tok   = (int*)(smem + HT_BYTES);       // [MT]

    const int tid   = threadIdx.x;
    const int wave  = tid >> 5;
    const int lane  = tid & 31;
    const int half  = lane >> 4;        // K-half selector
    const int lcol  = lane & 15;        // fragment row (A) / column (B,C)
    const int mbase = (wave & 1) * 16;  // this wave's M base (2 M-wave groups)
    const int wnb   = (wave >> 1) * 128;// this wave's N base (8 N-wave groups)

    int tileStart, segEnd;
    const __bf16 *W1, *W2;
    const float *B1, *B2;
    if (perm) {
        const int e  = blockIdx.y;
        const int s0 = offsets[e], s1 = offsets[e + 1];
        tileStart = s0 + (int)blockIdx.x * MT;
        if (tileStart >= s1) return;           // block-uniform exit
        segEnd = s1;
        W1 = W1t + (size_t)e * Hc * Hc;  B1 = B1base + (size_t)e * Hc;
        W2 = W2t + (size_t)e * Hc * Hc;  B2 = B2base + (size_t)e * Hc;
    } else {
        tileStart = (int)blockIdx.x * MT;
        segEnd = 1 << 30;
        W1 = W1t; B1 = B1base; W2 = W2t; B2 = B2base;
    }

    if (tid < MT) {
        const int idx = tileStart + tid;
        tok[tid] = perm ? ((idx < segEnd) ? perm[idx] : -1) : idx;
    }
    __syncthreads();

    // Per-lane gathered A row (pad rows clamp to 0; epilogue guard prevents stores)
    int trow = tok[mbase + lcol];
    if (trow < 0) trow = 0;
    const __bf16* xrow = Xb + (size_t)trow * Hc;

    // ---------------- Phase 1: Htile = relu(X @ W1 + b1) ----------------
    {
        v8f acc[8] = {};
        const int ka = half * 8;        // A low-half K offset within chunk
        const int kb = half * 8;        // B K offset within chunk
        for (int k0 = 0; k0 < Hc; k0 += 32) {
            const v8bf alo = *(const v8bf*)(xrow + k0 + ka);
            const v8bf ahi = *(const v8bf*)(xrow + k0 + 16 + ka);
            const v16bf afrag = make_frag(alo, ahi);

            if (k0 + 32 < Hc)
                __builtin_prefetch((const void*)(W1 + (size_t)(wnb + lane) * Hc + k0 + 32), 0, 1);

            // batch-issue all 16 B-fragment b128 loads, then run the WMMA chain
            v16bf bf[8];
            #pragma unroll
            for (int t = 0; t < 8; ++t) {
                const __bf16* wp = W1 + (size_t)(wnb + t * 16 + lcol) * Hc + k0 + kb;
                v8bf* dst = (v8bf*)&bf[t];
                dst[0] = *(const v8bf*)wp;
                dst[1] = *(const v8bf*)(wp + 16);
            }
            #pragma unroll
            for (int t = 0; t < 8; ++t)
                acc[t] = __builtin_amdgcn_wmma_f32_16x16x32_bf16(
                    false, afrag, false, bf[t], (short)0, acc[t], false, false);
        }

        // bias + relu -> bf16 LDS (C layout: m = mbase + r + 8*half, n = lcol)
        #pragma unroll
        for (int t = 0; t < 8; ++t) {
            const int n = wnb + t * 16 + lcol;
            const float bias = B1[n];
            #pragma unroll
            for (int r = 0; r < 8; ++r) {
                const int m = mbase + r + half * 8;
                Htile[(size_t)m * HS + n] = (__bf16)fmaxf(acc[t][r] + bias, 0.0f);
            }
        }
    }
    __syncthreads();

    // ---------------- Phase 2: Out rows = Htile @ W2 + b2 ----------------
    {
        v8f acc[8] = {};
        const int ka = half * 8;
        const int kb = half * 8;
        const __bf16* hrow = Htile + (size_t)(mbase + lcol) * HS;
        for (int k0 = 0; k0 < Hc; k0 += 32) {
            const v8bf alo = *(const v8bf*)(hrow + k0 + ka);
            const v8bf ahi = *(const v8bf*)(hrow + k0 + 16 + ka);
            const v16bf afrag = make_frag(alo, ahi);

            if (k0 + 32 < Hc)
                __builtin_prefetch((const void*)(W2 + (size_t)(wnb + lane) * Hc + k0 + 32), 0, 1);

            v16bf bf[8];
            #pragma unroll
            for (int t = 0; t < 8; ++t) {
                const __bf16* wp = W2 + (size_t)(wnb + t * 16 + lcol) * Hc + k0 + kb;
                v8bf* dst = (v8bf*)&bf[t];
                dst[0] = *(const v8bf*)wp;
                dst[1] = *(const v8bf*)(wp + 16);
            }
            #pragma unroll
            for (int t = 0; t < 8; ++t)
                acc[t] = __builtin_amdgcn_wmma_f32_16x16x32_bf16(
                    false, afrag, false, bf[t], (short)0, acc[t], false, false);
        }

        // epilogue: bias + scatter to gathered token rows
        #pragma unroll
        for (int t = 0; t < 8; ++t) {
            const int n = wnb + t * 16 + lcol;
            const float bias = B2[n];
            #pragma unroll
            for (int r = 0; r < 8; ++r) {
                const int m = mbase + r + half * 8;
                const int tk = tok[m];
                if (tk >= 0) {
                    float* op = Out + (size_t)tk * Hc + n;
                    const float v = acc[t][r] + bias;
                    if (accumulate) *op = *op + v;
                    else            *op = v;
                }
            }
        }
    }
}

extern "C" void kernel_launch(void* const* d_in, const int* in_sizes, int n_in,
                              void* d_out, int out_size, void* d_ws, size_t ws_size,
                              hipStream_t stream) {
    const float* x   = (const float*)d_in[0];
    const float* ue  = (const float*)d_in[1];
    const float* sw1 = (const float*)d_in[2];
    const float* sb1 = (const float*)d_in[3];
    const float* sw2 = (const float*)d_in[4];
    const float* sb2 = (const float*)d_in[5];
    const float* ew1 = (const float*)d_in[6];
    const float* eb1 = (const float*)d_in[7];
    const float* ew2 = (const float*)d_in[8];
    const float* eb2 = (const float*)d_in[9];
    const float* uw1 = (const float*)d_in[10];
    const float* ub1 = (const float*)d_in[11];
    const float* uw2 = (const float*)d_in[12];
    const float* ub2 = (const float*)d_in[13];
    float* out = (float*)d_out;

    const int Bn = in_sizes[0] / Hc;   // 16384 tokens

    // workspace layout: bf16 buffers first (16B aligned), then ints
    char* ws = (char*)d_ws;
    size_t off = 0;
    __bf16* xb   = (__bf16*)(ws + off); off += (size_t)Bn * Hc * 2;
    __bf16* e1t  = (__bf16*)(ws + off); off += (size_t)Ec * Hc * Hc * 2;
    __bf16* e2t  = (__bf16*)(ws + off); off += (size_t)Ec * Hc * Hc * 2;
    __bf16* u1t  = (__bf16*)(ws + off); off += (size_t)Hc * Hc * 2;
    __bf16* u2t  = (__bf16*)(ws + off); off += (size_t)Hc * Hc * 2;
    int* routes  = (int*)(ws + off);    off += (size_t)Bn * 4;
    int* offsets = (int*)(ws + off);    off += 16 * 4;
    int* perm    = (int*)(ws + off);

    // 0) one-time precision/layout conversion
    const int nx = Bn * Hc;
    moe_cvt_bf16_kernel<<<nx / (256 * 4), 256, 0, stream>>>(x, xb, nx);
    moe_transpose_cvt_kernel<<<dim3(32, 32, Ec), 256, 0, stream>>>(ew1, e1t);
    moe_transpose_cvt_kernel<<<dim3(32, 32, Ec), 256, 0, stream>>>(ew2, e2t);
    moe_transpose_cvt_kernel<<<dim3(32, 32, 1), 256, 0, stream>>>(uw1, u1t);
    moe_transpose_cvt_kernel<<<dim3(32, 32, 1), 256, 0, stream>>>(uw2, u2t);

    // 1) route
    moe_router_kernel<<<Bn / 8, 256, 0, stream>>>(ue, sw1, sb1, sw2, sb2, routes);
    // 2) group tokens by expert
    moe_bucketize_kernel<<<1, 256, 0, stream>>>(routes, offsets, perm, Bn);
    // 3) routed expert FFN (writes '='; every token covered exactly once)
    dim3 gexp(Bn / MT, Ec);
    moe_ffn_wmma_kernel<<<gexp, 512, SMEM_BYTES, stream>>>(xb, e1t, eb1, e2t, eb2, out,
                                                           perm, offsets, /*accumulate=*/0);
    // 4) shared user FFN accumulated on top (same stream => ordered)
    moe_ffn_wmma_kernel<<<Bn / MT, 512, SMEM_BYTES, stream>>>(xb, u1t, ub1, u2t, ub2, out,
                                                              nullptr, nullptr, /*accumulate=*/1);
}
